// MultiHeadedAttention_82815559401884
// MI455X (gfx1250) — compile-verified, matrix operand order checked by reading, not run on hardware
//
#include <hip/hip_runtime.h>

// MI455X / gfx1250, wave32. Compute-bound MHA (~137 GFLOP vs ~6us HBM @23.3TB/s):
// all matmuls on the WMMA pipe (v_wmma_f32_16x16x32_bf16, f32 accum).
// LDS tiles are fragment-major (contiguous K, rows 16B-aligned, odd-bank
// stride) so every WMMA fragment is 2x ds_load_b128. GEMM waves compute 32x64
// (8 wmma / 12 b128 per K-step). Softmax row-sums ride the matrix pipe
// (P @ ones); softmax is base-2 (v_exp_f32 is exp2) with mask as additive bias.

typedef __bf16 bf16_t;
typedef __attribute__((ext_vector_type(16))) __bf16 v16bf;
typedef __attribute__((ext_vector_type(8)))  __bf16 v8bf;
typedef __attribute__((ext_vector_type(8)))  float  v8f;

#define B_   4
#define S_   2048
#define D_   1024
#define H_   16
#define HD_  64
#define M_   (B_ * S_)   // 8192 rows

// ---------------- WMMA wrapper ----------------
__device__ static inline v8f wmma_bf16(v16bf a, v16bf b, v8f c) {
  return __builtin_amdgcn_wmma_f32_16x16x32_bf16(false, a, false, b,
                                                 (short)0, c, false, false);
}

// Fragment load from fragment-major storage [row][k] (row = m for A, n for B^T):
// regs 0-3 hold K = koff..koff+7 (contiguous), regs 4-7 hold K = koff+16..+23.
// Requires rows 16B-aligned (ld*2 % 16 == 0) -> two 128-bit loads per lane.
__device__ static inline v16bf frag_ld(const bf16_t* t, int ld, int lane, int kbase) {
  const int row  = lane & 15;
  const int koff = ((lane >> 4) << 3) + kbase;
  const bf16_t* p = t + row * ld + koff;
  v8bf lo = *(const v8bf*)(p);        // K = koff .. koff+7
  v8bf hi = *(const v8bf*)(p + 16);   // K = koff+16 .. koff+23
  return __builtin_shufflevector(lo, hi, 0, 1, 2, 3, 4, 5, 6, 7,
                                 8, 9, 10, 11, 12, 13, 14, 15);
}

__device__ static inline unsigned int pack2_bf16(float x, float y) {
#if __has_builtin(__builtin_amdgcn_cvt_pk_bf16_f32)
  auto p = __builtin_amdgcn_cvt_pk_bf16_f32(x, y);   // single v_cvt_pk_bf16_f32
  return __builtin_bit_cast(unsigned int, p);
#else
  unsigned short lo = __builtin_bit_cast(unsigned short, (bf16_t)x);
  unsigned short hi = __builtin_bit_cast(unsigned short, (bf16_t)y);
  return ((unsigned int)hi << 16) | lo;
#endif
}

// ---------------- GEMM: C[m,n] = sum_k A[m,k]*W[n,k] + bias[n] ----------------
// Block tile 128(M) x 128(N), K-step 32. 256 threads = 8 waves (4xM, 2xN);
// each wave computes 32x64: 2 A-frags x 4 B-frags = 8 wmma per K-step.
// A staged as As[m][k]; B staged transposed Bt[n][k] (coalesced from W rows).
template <bool A_BF16, bool HEADSPLIT>
__global__ __launch_bounds__(256) void gemm_wmma_kernel(
    const void* __restrict__ Ap, const float* __restrict__ W,
    const float* __restrict__ bias, void* __restrict__ outp) {
  __shared__ __align__(16) bf16_t As[128][40];   // 10 KB
  __shared__ __align__(16) bf16_t Bt[128][40];   // 10 KB

  const int tid   = threadIdx.x;
  const int lane  = tid & 31;
  const int wid   = tid >> 5;
  const int waveM = wid & 3;    // 4 waves along M, 32 rows each
  const int waveN = wid >> 2;   // 2 waves along N, 64 cols each
  const int m0    = blockIdx.x * 128;
  const int n0    = blockIdx.y * 128;

  const float*  Af = (const float*)Ap;
  const bf16_t* Ab = (const bf16_t*)Ap;

  v8f acc[2][4] = {};

  for (int k0 = 0; k0 < D_; k0 += 32) {
    // Stage A tile 128x32 as packed dwords (2048 pairs / 256 threads = 8 each)
#pragma unroll
    for (int i = 0; i < 8; ++i) {
      const int idx = tid + i * 256;
      const int r = idx >> 4, c2 = idx & 15;
      unsigned int w;
      if (A_BF16) {
        w = ((const unsigned int*)(Ab + (size_t)(m0 + r) * D_ + k0))[c2];
      } else {
        const float2 v = ((const float2*)(Af + (size_t)(m0 + r) * D_ + k0))[c2];
        w = pack2_bf16(v.x, v.y);
      }
      ((unsigned int*)&As[r][0])[c2] = w;
    }
    // Stage B^T tile 128x32 (coalesced along k within W rows)
#pragma unroll
    for (int i = 0; i < 8; ++i) {
      const int idx = tid + i * 256;
      const int n = idx >> 4, k2 = idx & 15;
      const float2 v = ((const float2*)(W + (size_t)(n0 + n) * D_ + k0))[k2];
      ((unsigned int*)&Bt[n][0])[k2] = pack2_bf16(v.x, v.y);
    }
    // Prefetch next K-step's rows (global_prefetch_b8)
    if (k0 + 32 < D_) {
      if (!A_BF16)
        __builtin_prefetch(Af + (size_t)(m0 + (tid & 127)) * D_ + k0 + 32, 0, 0);
      __builtin_prefetch(W + (size_t)(n0 + (tid & 127)) * D_ + k0 + 32, 0, 0);
    }
    __syncthreads();

    const v16bf af0 = frag_ld(&As[(waveM * 2 + 0) * 16][0], 40, lane, 0);
    const v16bf af1 = frag_ld(&As[(waveM * 2 + 1) * 16][0], 40, lane, 0);
#pragma unroll
    for (int t = 0; t < 4; ++t) {
      const v16bf bfrag = frag_ld(&Bt[waveN * 64 + t * 16][0], 40, lane, 0);
      acc[0][t] = wmma_bf16(af0, bfrag, acc[0][t]);
      acc[1][t] = wmma_bf16(af1, bfrag, acc[1][t]);
    }
    __syncthreads();
  }

  // C layout: lanes 0-15 -> rows 0-7, lanes 16-31 -> rows 8-15; col = lane%16
  const int rbase = (lane >> 4) << 3;
  const int cl    = lane & 15;
#pragma unroll
  for (int mt = 0; mt < 2; ++mt) {
#pragma unroll
    for (int t = 0; t < 4; ++t) {
      const int col = n0 + waveN * 64 + t * 16 + cl;
      const float bv = bias[col];
#pragma unroll
      for (int r = 0; r < 8; ++r) {
        const int row = m0 + (waveM * 2 + mt) * 16 + rbase + r;
        const float v = acc[mt][t][r] + bv;
        if (HEADSPLIT) {
          const int b = row >> 11, s = row & (S_ - 1);   // S = 2048
          const int h = col >> 6,  d = col & (HD_ - 1);  // HD = 64
          ((bf16_t*)outp)[(((size_t)b * H_ + h) * S_ + s) * HD_ + d] = (bf16_t)v;
        } else {
          ((float*)outp)[(size_t)row * D_ + col] = v;
        }
      }
    }
  }
}

// ---------------- Flash attention per (b,h) ----------------
// 128 threads = 4 waves; each wave owns 16 query rows (block covers 64).
// 64-key slabs: QK^T (8 wmma), base-2 online softmax (row max via shfl_xor
// within 16-lane halves, row sums via P@ones WMMA), PV (8 wmma).
__global__ __launch_bounds__(128) void attn_kernel(
    const bf16_t* __restrict__ Qh, const bf16_t* __restrict__ Kh,
    const bf16_t* __restrict__ Vh, const unsigned char* __restrict__ mask,
    bf16_t* __restrict__ ctxb) {
  __shared__ __align__(16) bf16_t Ks[64][72];      // [key][d], 144B rows
  __shared__ __align__(16) bf16_t Vt[64][72];      // [d][key] transposed
  __shared__ __align__(16) bf16_t Ps[4][16][72];   // per-wave P relayout tile

  const int tid  = threadIdx.x;
  const int lane = tid & 31;
  const int wid  = tid >> 5;
  const int bh   = blockIdx.x;
  const int b    = bh >> 4;   // H = 16
  const int h    = bh & 15;
  const int q0   = blockIdx.y * 64 + wid * 16;

  // Q fragments live in registers all kernel (HD=64 -> 2 K-steps, 2x b128 each)
  const bf16_t* Qbase = Qh + ((size_t)bh * S_ + q0) * HD_;
  const v16bf qa0 = frag_ld(Qbase, HD_, lane, 0);
  const v16bf qa1 = frag_ld(Qbase, HD_, lane, 32);

  v16bf onesf;
#pragma unroll
  for (int j = 0; j < 16; ++j) onesf[j] = (bf16_t)1.0f;

  v8f ctx[4] = {};
  float mrow[8], lrow[8];
#pragma unroll
  for (int r = 0; r < 8; ++r) { mrow[r] = -1e30f; lrow[r] = 0.f; }

  const unsigned char* mk = mask + (size_t)b * S_;
  const float scale2 = 0.125f * 1.44269504089f;  // (1/sqrt(64)) * log2(e)
  const int cl    = lane & 15;
  const int rbase = (lane >> 4) << 3;

  for (int s0 = 0; s0 < S_; s0 += 64) {
    const bf16_t* Ksrc = Kh + ((size_t)bh * S_ + s0) * HD_;
    const bf16_t* Vsrc = Vh + ((size_t)bh * S_ + s0) * HD_;
    // Stage 64x64 K slab ([key][d]) and transposed V slab ([d][key]).
#pragma unroll
    for (int i = 0; i < 16; ++i) {
      const int idx = tid + i * 128;          // 2048 dwords
      const int key = idx >> 5, dp = idx & 31;
      ((unsigned int*)&Ks[key][0])[dp] =
          ((const unsigned int*)(Ksrc + key * HD_))[dp];
      const unsigned int w = ((const unsigned int*)(Vsrc + key * HD_))[dp];
      const int d = dp * 2;
      Vt[d][key]     = __builtin_bit_cast(bf16_t, (unsigned short)(w & 0xffff));
      Vt[d + 1][key] = __builtin_bit_cast(bf16_t, (unsigned short)(w >> 16));
    }
    __syncthreads();

    // scores = Q @ K^T : 4 key n-tiles x 2 K-steps
    v8f sc[4];
    float mbias[4];
#pragma unroll
    for (int nt = 0; nt < 4; ++nt) {
      v8f s = {};
      s = wmma_bf16(qa0, frag_ld(&Ks[nt * 16][0], 72, lane, 0),  s);
      s = wmma_bf16(qa1, frag_ld(&Ks[nt * 16][0], 72, lane, 32), s);
      sc[nt] = s;
      mbias[nt] = (mk[s0 + nt * 16 + cl] != 0) ? 0.0f : -1e30f;
    }

    float corrv[8];
#pragma unroll
    for (int r = 0; r < 8; ++r) {
      float a0 = sc[0][r] * scale2 + mbias[0];
      float a1 = sc[1][r] * scale2 + mbias[1];
      float a2 = sc[2][r] * scale2 + mbias[2];
      float a3 = sc[3][r] * scale2 + mbias[3];
      // row max over 64 columns: 4-way local, then across the 16-lane half
      float lm = fmaxf(fmaxf(a0, a1), fmaxf(a2, a3));
#pragma unroll
      for (int o = 8; o > 0; o >>= 1) lm = fmaxf(lm, __shfl_xor(lm, o, 16));
      const float mn = fmaxf(mrow[r], lm);
      corrv[r] = exp2f(mrow[r] - mn);
      mrow[r] = mn;
      const int row = rbase + r;
      Ps[wid][row][cl]      = (bf16_t)exp2f(a0 - mn);
      Ps[wid][row][cl + 16] = (bf16_t)exp2f(a1 - mn);
      Ps[wid][row][cl + 32] = (bf16_t)exp2f(a2 - mn);
      Ps[wid][row][cl + 48] = (bf16_t)exp2f(a3 - mn);
#pragma unroll
      for (int t = 0; t < 4; ++t) ctx[t][r] *= corrv[r];
    }

    // P fragments (16x64 -> 2 K-steps)
    const v16bf pf0 = frag_ld(&Ps[wid][0][0], 72, lane, 0);
    const v16bf pf1 = frag_ld(&Ps[wid][0][0], 72, lane, 32);

    // Row sums via matrix pipe: rs = P @ ones (replicated across columns)
    v8f rs = {};
    rs = wmma_bf16(pf0, onesf, rs);
    rs = wmma_bf16(pf1, onesf, rs);
#pragma unroll
    for (int r = 0; r < 8; ++r) lrow[r] = lrow[r] * corrv[r] + rs[r];

    // ctx += P @ V : B-frags from Vt[d][key] (n = d, K = key, 2 K-steps)
#pragma unroll
    for (int t = 0; t < 4; ++t) {
      ctx[t] = wmma_bf16(pf0, frag_ld(&Vt[t * 16][0], 72, lane, 0),  ctx[t]);
      ctx[t] = wmma_bf16(pf1, frag_ld(&Vt[t * 16][0], 72, lane, 32), ctx[t]);
    }
    __syncthreads();
  }

  // Normalize and store context bf16 [B,S,D] (col = h*HD + d)
#pragma unroll
  for (int r = 0; r < 8; ++r) {
    const float inv = 1.0f / lrow[r];
    const int qrow = q0 + rbase + r;
#pragma unroll
    for (int t = 0; t < 4; ++t) {
      const int d = t * 16 + cl;
      ctxb[((size_t)b * S_ + qrow) * D_ + h * HD_ + d] = (bf16_t)(ctx[t][r] * inv);
    }
  }
}

// ---------------- host launcher ----------------
extern "C" void kernel_launch(void* const* d_in, const int* in_sizes, int n_in,
                              void* d_out, int out_size, void* d_ws, size_t ws_size,
                              hipStream_t stream) {
  (void)in_sizes; (void)n_in; (void)out_size; (void)ws_size;

  const float* query = (const float*)d_in[0];
  const float* key   = (const float*)d_in[1];
  const float* value = (const float*)d_in[2];
  const unsigned char* mask = (const unsigned char*)d_in[3];  // jax bool = 1 byte
  const float* Wq = (const float*)d_in[4];
  const float* bq = (const float*)d_in[5];
  const float* Wk = (const float*)d_in[6];
  const float* bk = (const float*)d_in[7];
  const float* Wv = (const float*)d_in[8];
  const float* bv = (const float*)d_in[9];
  const float* Wo = (const float*)d_in[10];
  const float* bo = (const float*)d_in[11];

  // Workspace: Qh/Kh/Vh bf16 [B,H,S,HD] (16 MiB each) + ctx bf16 [B,S,D]
  bf16_t* Qh  = (bf16_t*)d_ws;
  bf16_t* Kh  = Qh + (size_t)M_ * D_;
  bf16_t* Vh  = Kh + (size_t)M_ * D_;
  bf16_t* Ctx = Vh + (size_t)M_ * D_;

  dim3 gGemm(M_ / 128, D_ / 128);  // 64 x 8 blocks
  gemm_wmma_kernel<false, true><<<gGemm, 256, 0, stream>>>(query, Wq, bq, Qh);
  gemm_wmma_kernel<false, true><<<gGemm, 256, 0, stream>>>(key,   Wk, bk, Kh);
  gemm_wmma_kernel<false, true><<<gGemm, 256, 0, stream>>>(value, Wv, bv, Vh);

  dim3 gAttn(B_ * H_, S_ / 64);   // 64 x 32 blocks
  attn_kernel<<<gAttn, 128, 0, stream>>>(Qh, Kh, Vh, mask, Ctx);

  gemm_wmma_kernel<true, false><<<gGemm, 256, 0, stream>>>(Ctx, Wo, bo, d_out);
}
